// Encoder_73976516706310
// MI455X (gfx1250) — matrix operand order repaired and checked by reading, not compiled, not run on hardware
//
#include <hip/hip_runtime.h>
#include <hip/hip_bf16.h>

typedef __bf16 bf16;
typedef __attribute__((ext_vector_type(16))) __bf16 v16bf;
typedef __attribute__((ext_vector_type(8)))  float  v8f;

union FragAB { v16bf v; uint4 u[2]; };

#define EP_BIAS  0
#define EP_RELU  1
#define EP_SCALE 2
#define EP_PE    3
#define EP_RESID 4

// C[M,N] = A[M,K] * B[K,N] (+ epilogue), A bf16 row-major, BT = B^T bf16 row-major [N,K].
// Block tile 128x128, K-step 32, double-buffered LDS, one __syncthreads per K-step.
// 256 threads = 8 waves in 4x2 grid; each wave computes 32x64 (2x4 WMMA tiles, 8 WMMA/K-step).
// Loop body order: compute(cur) then stage(nxt) direct global->LDS, so no staging registers
// live across the WMMA block (avoids spills); the scheduler hoists the global loads early.
template<int EP, bool OUTF32>
__global__ void __launch_bounds__(256) gemm_bf16_wmma(
    const bf16* __restrict__ A,  long sAb,
    const bf16* __restrict__ BT, long sBb,
    void* __restrict__ Cv,       long sCb,
    int K, int lda, int ldb, int ldc,
    const float* __restrict__ bias,
    const float* __restrict__ pe, int pe_mod,
    const bf16* __restrict__ resid,
    float scale)
{
    __shared__ bf16 sA[2][128 * 32];
    __shared__ bf16 sB[2][128 * 32];

    const int tid  = threadIdx.x;
    const int lane = tid & 31;
    const int wave = tid >> 5;
    const int wm   = wave >> 1;   // 0..3 : rows of 32
    const int wn   = wave & 1;    // 0..1 : cols of 64
    const int m0   = blockIdx.y * 128;
    const int n0   = blockIdx.x * 128;
    const int z    = blockIdx.z;

    const bf16* Ab = A  + (long)z * sAb;
    const bf16* Bb = BT + (long)z * sBb;

    // staging: 128x32 halves = 512 uint4 per tile; each thread moves 2 for A, 2 for B
    const int r0t = tid >> 2;         // 0..63
    const int c0t = (tid & 3) << 3;   // 0,8,16,24

    v8f acc[2][4] = {};
    const int nk = K >> 5;

    // prologue: stage K-tile 0 into buffer 0
    #pragma unroll
    for (int i = 0; i < 2; ++i) {
        const int r = r0t + i * 64;
        *(uint4*)&sA[0][r * 32 + c0t] = *(const uint4*)&Ab[(long)(m0 + r) * lda + c0t];
        *(uint4*)&sB[0][r * 32 + c0t] = *(const uint4*)&Bb[(long)(n0 + r) * ldb + c0t];
    }
    __syncthreads();

    const int mrow  = lane & 15;
    const int koffA = (lane >> 4) << 3;   // A frag: lanes>=16 take K 8-15 / 24-31 (interleaved)
    const int koffB = (lane >> 4) << 3;   // B frag: lanes>=16 take K 8-15 / 24-31 (interleaved)

    for (int kt = 0; kt < nk; ++kt) {
        const int cur = kt & 1;

        // fragments from current buffer + 8 WMMA
        FragAB fa[2], fb[4];
        #pragma unroll
        for (int t = 0; t < 2; ++t) {
            const int ra = wm * 32 + t * 16 + mrow;
            fa[t].u[0] = *(const uint4*)&sA[cur][ra * 32 + koffA];
            fa[t].u[1] = *(const uint4*)&sA[cur][ra * 32 + koffA + 16];
        }
        #pragma unroll
        for (int t = 0; t < 4; ++t) {
            const int rb = wn * 64 + t * 16 + mrow;
            fb[t].u[0] = *(const uint4*)&sB[cur][rb * 32 + koffB];
            fb[t].u[1] = *(const uint4*)&sB[cur][rb * 32 + koffB + 16];
        }
        #pragma unroll
        for (int tm = 0; tm < 2; ++tm)
            #pragma unroll
            for (int tn = 0; tn < 4; ++tn)
                acc[tm][tn] = __builtin_amdgcn_wmma_f32_16x16x32_bf16(
                    false, fa[tm].v, false, fb[tn].v,
                    (short)0, acc[tm][tn], false, false);

        // stage next K-tile directly global -> LDS (other buffer)
        if (kt + 1 < nk) {
            const int nxt = cur ^ 1;
            const int k0  = (kt + 1) << 5;
            #pragma unroll
            for (int i = 0; i < 2; ++i) {
                const int r = r0t + i * 64;
                *(uint4*)&sA[nxt][r * 32 + c0t] =
                    *(const uint4*)&Ab[(long)(m0 + r) * lda + k0 + c0t];
                *(uint4*)&sB[nxt][r * 32 + c0t] =
                    *(const uint4*)&Bb[(long)(n0 + r) * ldb + k0 + c0t];
            }
        }
        __syncthreads();
    }

    // Epilogue: lane holds N = lane&15; acc element r is row M = r + (lane>=16)*8
    const int mrow8 = (lane >> 4) * 8;
    #pragma unroll
    for (int tm = 0; tm < 2; ++tm) {
        #pragma unroll
        for (int tn = 0; tn < 4; ++tn) {
            const int n = n0 + wn * 64 + tn * 16 + (lane & 15);
            #pragma unroll
            for (int r = 0; r < 8; ++r) {
                const int m = m0 + wm * 32 + tm * 16 + mrow8 + r;
                float v = acc[tm][tn][r];
                if (EP == EP_SCALE) v *= scale;
                if (EP == EP_BIAS || EP == EP_RELU || EP == EP_PE) v += bias[n];
                if (EP == EP_PE)    v += pe[(long)(m & (pe_mod - 1)) * ldc + n]; // pe_mod is pow2
                if (EP == EP_RELU)  v = v > 0.f ? v : 0.f;
                if (EP == EP_RESID) v += (float)resid[(long)z * sCb + (long)m * ldc + n];
                const long coff = (long)z * sCb + (long)m * ldc + n;
                if (OUTF32) ((float*)Cv)[coff] = v;
                else        ((bf16*)Cv)[coff]  = (bf16)v;
            }
        }
    }
}

// Row softmax over bf16 scores, one 256-thread block per row of length 2048.
__global__ void __launch_bounds__(256) softmax_rows(bf16* __restrict__ S, int rowlen)
{
    __shared__ float row[2048];
    __shared__ float red[256];
    const long base = (long)blockIdx.x * rowlen;
    const int tid = threadIdx.x;

    float lmax = -3.4e38f;
    for (int i = tid; i < rowlen; i += 256) {
        float v = (float)S[base + i];
        row[i] = v;
        lmax = fmaxf(lmax, v);
    }
    red[tid] = lmax;
    __syncthreads();
    for (int s = 128; s > 0; s >>= 1) {
        if (tid < s) red[tid] = fmaxf(red[tid], red[tid + s]);
        __syncthreads();
    }
    const float mx = red[0];
    __syncthreads();

    float lsum = 0.f;
    for (int i = tid; i < rowlen; i += 256) {
        float e = __expf(row[i] - mx);
        row[i] = e;
        lsum += e;
    }
    red[tid] = lsum;
    __syncthreads();
    for (int s = 128; s > 0; s >>= 1) {
        if (tid < s) red[tid] += red[tid + s];
        __syncthreads();
    }
    const float inv = 1.f / red[0];
    for (int i = tid; i < rowlen; i += 256)
        S[base + i] = (bf16)(row[i] * inv);
}

// LayerNorm over rows of length D=512 (2 elems/thread), one block per row.
__global__ void __launch_bounds__(256) layernorm_rows(
    const bf16* __restrict__ H,
    const float* __restrict__ gamma, const float* __restrict__ beta,
    bf16* __restrict__ O, int D)
{
    __shared__ float s1[256], s2[256];
    const long base = (long)blockIdx.x * D;
    const int tid = threadIdx.x;
    const float v0 = (float)H[base + tid];
    const float v1 = (float)H[base + tid + 256];
    s1[tid] = v0 + v1;
    s2[tid] = v0 * v0 + v1 * v1;
    __syncthreads();
    for (int s = 128; s > 0; s >>= 1) {
        if (tid < s) { s1[tid] += s1[tid + s]; s2[tid] += s2[tid + s]; }
        __syncthreads();
    }
    const float mu  = s1[0] / (float)D;
    const float var = s2[0] / (float)D - mu * mu;
    const float rs  = rsqrtf(var + 1e-5f);
    O[base + tid]       = (bf16)((v0 - mu) * rs * gamma[tid]       + beta[tid]);
    O[base + tid + 256] = (bf16)((v1 - mu) * rs * gamma[tid + 256] + beta[tid + 256]);
}

// Sinusoidal positional encoding, f32 [S,D].
__global__ void build_pe(float* __restrict__ pe, int S, int D)
{
    long i = (long)blockIdx.x * blockDim.x + threadIdx.x;
    if (i >= (long)S * D) return;
    int s = (int)(i / D), d = (int)(i % D);
    int k = d >> 1;
    float div = __expf(-logf(10000.f) * (2.f * k) / (float)D);
    float a = (float)s * div;
    pe[i] = (d & 1) ? cosf(a) : sinf(a);
}

// f32 [K,N] -> bf16 transposed [N,K] (dims multiples of 32).
__global__ void __launch_bounds__(256) transpose_f32_bf16(
    const float* __restrict__ W, bf16* __restrict__ WT, int K, int N)
{
    __shared__ float tile[32][33];
    const int n0 = blockIdx.x * 32, k0 = blockIdx.y * 32;
    const int tx = threadIdx.x & 31, ty = threadIdx.x >> 5;
    #pragma unroll
    for (int i = 0; i < 32; i += 8)
        tile[ty + i][tx] = W[(long)(k0 + ty + i) * N + n0 + tx];
    __syncthreads();
    #pragma unroll
    for (int i = 0; i < 32; i += 8)
        WT[(long)(n0 + ty + i) * K + k0 + tx] = (bf16)tile[tx][ty + i];
}

// Batched bf16 transpose: per batch, in [R,C] -> out [C,R].
__global__ void __launch_bounds__(256) transpose_bf16(
    const bf16* __restrict__ in, bf16* __restrict__ out,
    int R, int C, long sIn, long sOut)
{
    __shared__ bf16 tile[32][33];
    const bf16* ib = in  + (long)blockIdx.z * sIn;
    bf16*       ob = out + (long)blockIdx.z * sOut;
    const int c0 = blockIdx.x * 32, r0 = blockIdx.y * 32;
    const int tx = threadIdx.x & 31, ty = threadIdx.x >> 5;
    #pragma unroll
    for (int i = 0; i < 32; i += 8)
        tile[ty + i][tx] = ib[(long)(r0 + ty + i) * C + c0 + tx];
    __syncthreads();
    #pragma unroll
    for (int i = 0; i < 32; i += 8)
        ob[(long)(c0 + ty + i) * R + r0 + tx] = tile[tx][ty + i];
}

__global__ void f32_to_bf16(const float* __restrict__ in, bf16* __restrict__ out, long n)
{
    long i = (long)blockIdx.x * 256 + threadIdx.x;
    if (i < n) out[i] = (bf16)in[i];
}

extern "C" void kernel_launch(void* const* d_in, const int* in_sizes, int n_in,
                              void* d_out, int out_size, void* d_ws, size_t ws_size,
                              hipStream_t stream)
{
    const int B = 8, S = 2048, P = 768, D = 512, F = 2048, L = 6;
    const int M = B * S;   // 16384
    const float RSCALE = 0.044194173824159216f;  // 1/sqrt(512)

    const float* tokens = (const float*)d_in[0];
    const float* W_emb  = (const float*)d_in[1];
    const float* b_emb  = (const float*)d_in[2];
    const float* Wq     = (const float*)d_in[3];
    const float* bq     = (const float*)d_in[4];
    const float* Wk     = (const float*)d_in[5];
    const float* bk     = (const float*)d_in[6];
    const float* Wv     = (const float*)d_in[7];
    const float* bv     = (const float*)d_in[8];
    const float* gamma  = (const float*)d_in[9];
    const float* beta   = (const float*)d_in[10];
    const float* W1     = (const float*)d_in[11];
    const float* b1     = (const float*)d_in[12];
    const float* W2     = (const float*)d_in[13];
    const float* b2     = (const float*)d_in[14];

    size_t off = 0;
    auto carve = [&](size_t bytes) -> void* {
        void* p = (void*)((char*)d_ws + off);
        off += (bytes + 255) & ~(size_t)255;
        return p;
    };
    bf16*  tokB  = (bf16*) carve((size_t)M * P * 2);
    bf16*  WembT = (bf16*) carve((size_t)D * P * 2);
    bf16*  WqT   = (bf16*) carve((size_t)L * D * D * 2);
    bf16*  WkT   = (bf16*) carve((size_t)L * D * D * 2);
    bf16*  WvT   = (bf16*) carve((size_t)L * D * D * 2);
    bf16*  W1T   = (bf16*) carve((size_t)L * F * D * 2);   // [F,D] per layer
    bf16*  W2T   = (bf16*) carve((size_t)L * D * F * 2);   // [D,F] per layer
    float* pe    = (float*)carve((size_t)S * D * 4);
    bf16*  X     = (bf16*) carve((size_t)M * D * 2);
    bf16*  Q     = (bf16*) carve((size_t)M * D * 2);
    bf16*  Km    = (bf16*) carve((size_t)M * D * 2);
    bf16*  V     = (bf16*) carve((size_t)M * D * 2);
    bf16*  VT    = (bf16*) carve((size_t)M * D * 2);       // per-batch [D,S]
    bf16*  SC    = (bf16*) carve((size_t)B * S * S * 2);   // scores/probs
    bf16*  H     = (bf16*) carve((size_t)M * D * 2);
    bf16*  Xn    = (bf16*) carve((size_t)M * D * 2);
    bf16*  mid   = (bf16*) carve((size_t)M * F * 2);
    (void)ws_size; (void)n_in; (void)in_sizes; (void)out_size;

    // --- one-time prep ---
    f32_to_bf16<<<((long)M * P + 255) / 256, 256, 0, stream>>>(tokens, tokB, (long)M * P);
    transpose_f32_bf16<<<dim3(D / 32, P / 32), 256, 0, stream>>>(W_emb, WembT, P, D);
    for (int l = 0; l < L; ++l) {
        transpose_f32_bf16<<<dim3(D / 32, D / 32), 256, 0, stream>>>(
            Wq + (size_t)l * D * D, WqT + (size_t)l * D * D, D, D);
        transpose_f32_bf16<<<dim3(D / 32, D / 32), 256, 0, stream>>>(
            Wk + (size_t)l * D * D, WkT + (size_t)l * D * D, D, D);
        transpose_f32_bf16<<<dim3(D / 32, D / 32), 256, 0, stream>>>(
            Wv + (size_t)l * D * D, WvT + (size_t)l * D * D, D, D);
        transpose_f32_bf16<<<dim3(F / 32, D / 32), 256, 0, stream>>>(
            W1 + (size_t)l * D * F, W1T + (size_t)l * F * D, D, F);
        transpose_f32_bf16<<<dim3(D / 32, F / 32), 256, 0, stream>>>(
            W2 + (size_t)l * F * D, W2T + (size_t)l * D * F, F, D);
    }
    build_pe<<<((long)S * D + 255) / 256, 256, 0, stream>>>(pe, S, D);

    // --- embedding: X = tokens @ W_emb + b_emb + PE ---
    gemm_bf16_wmma<EP_PE, false><<<dim3(D / 128, M / 128, 1), 256, 0, stream>>>(
        tokB, 0, WembT, 0, X, 0, P, P, P, D, b_emb, pe, S, nullptr, 1.f);

    for (int l = 0; l < L; ++l) {
        const bf16* wq = WqT + (size_t)l * D * D;
        const bf16* wk = WkT + (size_t)l * D * D;
        const bf16* wv = WvT + (size_t)l * D * D;
        const bf16* w1 = W1T + (size_t)l * F * D;
        const bf16* w2 = W2T + (size_t)l * D * F;

        // Q/K/V projections
        gemm_bf16_wmma<EP_BIAS, false><<<dim3(D / 128, M / 128, 1), 256, 0, stream>>>(
            X, 0, wq, 0, Q, 0, D, D, D, D, bq + (size_t)l * D, nullptr, 1, nullptr, 1.f);
        gemm_bf16_wmma<EP_BIAS, false><<<dim3(D / 128, M / 128, 1), 256, 0, stream>>>(
            X, 0, wk, 0, Km, 0, D, D, D, D, bk + (size_t)l * D, nullptr, 1, nullptr, 1.f);
        gemm_bf16_wmma<EP_BIAS, false><<<dim3(D / 128, M / 128, 1), 256, 0, stream>>>(
            X, 0, wv, 0, V, 0, D, D, D, D, bv + (size_t)l * D, nullptr, 1, nullptr, 1.f);

        // V^T per batch for the A@V GEMM's B operand
        transpose_bf16<<<dim3(D / 32, S / 32, B), 256, 0, stream>>>(
            V, VT, S, D, (long)S * D, (long)D * S);

        // scores = Q @ K^T / sqrt(D)   (B^T operand = K itself, row-major [S,D])
        gemm_bf16_wmma<EP_SCALE, false><<<dim3(S / 128, S / 128, B), 256, 0, stream>>>(
            Q, (long)S * D, Km, (long)S * D, SC, (long)S * S,
            D, D, D, S, nullptr, nullptr, 1, nullptr, RSCALE);

        softmax_rows<<<B * S, 256, 0, stream>>>(SC, S);

        // H = A @ V + X   (residual fused)
        gemm_bf16_wmma<EP_RESID, false><<<dim3(D / 128, S / 128, B), 256, 0, stream>>>(
            SC, (long)S * S, VT, (long)D * S, H, (long)S * D,
            S, S, S, D, nullptr, nullptr, 1, X, 1.f);

        layernorm_rows<<<M, 256, 0, stream>>>(
            H, gamma + (size_t)l * D, beta + (size_t)l * D, Xn, D);

        // FFN
        gemm_bf16_wmma<EP_RELU, false><<<dim3(F / 128, M / 128, 1), 256, 0, stream>>>(
            Xn, 0, w1, 0, mid, 0, D, D, D, F, b1 + (size_t)l * F, nullptr, 1, nullptr, 1.f);

        if (l == L - 1) {
            gemm_bf16_wmma<EP_BIAS, true><<<dim3(D / 128, M / 128, 1), 256, 0, stream>>>(
                mid, 0, w2, 0, d_out, 0, F, F, F, D,
                b2 + (size_t)l * D, nullptr, 1, nullptr, 1.f);
        } else {
            gemm_bf16_wmma<EP_BIAS, false><<<dim3(D / 128, M / 128, 1), 256, 0, stream>>>(
                mid, 0, w2, 0, X, 0, F, F, F, D,
                b2 + (size_t)l * D, nullptr, 1, nullptr, 1.f);
        }
    }
}
